// CA_DWT_SIMTransformerBlock_90546500534957
// MI455X (gfx1250) — matrix operand order repaired and checked by reading, not compiled, not run on hardware
//
#include <hip/hip_runtime.h>
#include <hip/hip_bf16.h>
#include <math.h>

// ---------------- types ----------------
typedef __bf16 bf16;
typedef __attribute__((ext_vector_type(16))) __bf16 v16bf;
typedef __attribute__((ext_vector_type(8)))  __bf16 v8bf;
typedef __attribute__((ext_vector_type(8)))  float  v8f;
typedef __attribute__((ext_vector_type(8)))  float  v8fv;

#define CB   192
#define BB   2
#define HH   128
#define WW   128
#define NN   (HH*WW)
#define NHEAD 6
#define HD   32
#define NW2  64
#define MLPH 768
#define ROWS (BB*NN)               // 32768
#define LAMBDA_INIT 0.3555090676f  // 0.8 - 0.6*exp(-0.3)
#define QSCALE 0.1767766953f       // 32^-0.5

__device__ __forceinline__ v16bf cat8(v8bf lo, v8bf hi){
  v16bf r;
#pragma unroll
  for (int i = 0; i < 8; ++i){ r[i] = lo[i]; r[i+8] = hi[i]; }
  return r;
}

__device__ __forceinline__ float gelu_f(float x){
  float x3 = x*x*x;
  return 0.5f*x*(1.0f + tanhf(0.7978845608028654f*(x + 0.044715f*x3)));
}

// ---------------- small prep kernels ----------------
// W: K x N (row major, f32)  ->  Bt: N x K (row major, bf16)
__global__ void transpose_bf_kernel(const float* __restrict__ W, bf16* __restrict__ Bt,
                                    int K, int N){
  int idx = blockIdx.x*blockDim.x + threadIdx.x;
  if (idx >= K*N) return;
  int n = idx / K, k = idx % K;
  Bt[idx] = (bf16)W[(size_t)k*N + n];
}

// conv weight (O=192, I=192, 3, 3) -> Wt[tap][o][i] bf16
__global__ void convw_prep_kernel(const float* __restrict__ W, bf16* __restrict__ Wt){
  int idx = blockIdx.x*blockDim.x + threadIdx.x;
  if (idx >= 9*CB*CB) return;
  int tap = idx / (CB*CB);
  int rem = idx % (CB*CB);
  int o = rem / CB, i = rem % CB;
  Wt[idx] = (bf16)W[(size_t)(o*CB + i)*9 + tap];
}

// bias_tab[h][tq][tk] = rpb[relidx(tq,tk)][h]
__global__ void biastab_kernel(const float* __restrict__ rpb, float* __restrict__ bt){
  int idx = blockIdx.x*blockDim.x + threadIdx.x;
  if (idx >= NHEAD*NW2*NW2) return;
  int h = idx / (NW2*NW2);
  int rem = idx % (NW2*NW2);
  int tq = rem >> 6, tk = rem & 63;
  int i1 = tq >> 3, j1 = tq & 7, i2 = tk >> 3, j2 = tk & 7;
  int rid = (i1 - i2 + 7)*15 + (j1 - j2 + 7);
  bt[idx] = rpb[(size_t)rid*NHEAD + h];
}

// ---------------- layernorm (one wave per 192-row) ----------------
__global__ void layernorm_kernel(const float* __restrict__ in,
                                 const float* __restrict__ g, const float* __restrict__ b,
                                 float* __restrict__ out_f, bf16* __restrict__ out_b,
                                 float postscale, int rows){
  int wid = (blockIdx.x*blockDim.x + threadIdx.x) >> 5;
  int lane = threadIdx.x & 31;
  if (wid >= rows) return;
  const float* p = in + (size_t)wid*CB;
  float v[6];
  float s = 0.f;
#pragma unroll
  for (int i = 0; i < 6; ++i){ v[i] = p[lane + i*32]; s += v[i]; }
#pragma unroll
  for (int m = 1; m < 32; m <<= 1) s += __shfl_xor(s, m, 32);
  float mean = s * (1.0f/CB);
  float vs = 0.f;
#pragma unroll
  for (int i = 0; i < 6; ++i){ float d = v[i]-mean; vs += d*d; }
#pragma unroll
  for (int m = 1; m < 32; m <<= 1) vs += __shfl_xor(vs, m, 32);
  float inv = rsqrtf(vs * (1.0f/CB) + 1e-5f);
#pragma unroll
  for (int i = 0; i < 6; ++i){
    int c = lane + i*32;
    float o = ((v[i]-mean)*inv*g[c] + b[c]) * postscale;
    if (out_f) out_f[(size_t)wid*CB + c] = o;
    if (out_b) out_b[(size_t)wid*CB + c] = (bf16)o;
  }
}

// ---------------- geo_e = xn + geo_w*(pf @ Wgeo + bgeo)  (K=3, elementwise) ----------------
__global__ void geo_kernel(const float* __restrict__ xn, const float* __restrict__ pf,
                           const float* __restrict__ Wgeo, const float* __restrict__ bgeo,
                           const float* __restrict__ geow_p, bf16* __restrict__ out){
  size_t idx = (size_t)blockIdx.x*blockDim.x + threadIdx.x;
  if (idx >= (size_t)ROWS*CB) return;
  int c = (int)(idx % CB);
  size_t row = idx / CB;
  float gw = geow_p[0];
  float v = pf[row*3+0]*Wgeo[c] + pf[row*3+1]*Wgeo[CB + c] + pf[row*3+2]*Wgeo[2*CB + c] + bgeo[c];
  out[idx] = (bf16)(xn[idx] + gw*v);
}

// ---------------- generic WMMA GEMM: out = A(MxK) @ Bt^T(NxK) + bias, epilogues ----------------
// epmode: 0: out_f = v
//         1: out_bf = bf16(alpha*v)
//         2: out_bf = bf16(addsrc + (*scale_ptr)*v)
//         3: out_bf = bf16(gelu(v))
//         4: out_f  = addsrc + v
template <int AF32>
__global__ void gemm_wmma_kernel(const void* __restrict__ Aptr, const bf16* __restrict__ Bt,
                                 const float* __restrict__ bias, const float* __restrict__ addsrc,
                                 const float* __restrict__ scale_ptr,
                                 bf16* __restrict__ out_bf, float* __restrict__ out_f,
                                 int M, int K, int N, int epmode, float alpha){
  int gw = (blockIdx.x*blockDim.x + threadIdx.x) >> 5;
  int lane = threadIdx.x & 31;
  int tilesN = N >> 4;
  int ntiles = (M >> 4) * tilesN;
  if (gw >= ntiles) return;
  int tm = gw / tilesN, tn = gw % tilesN;
  int half = lane >> 4, l16 = lane & 15;
  int arow = tm*16 + l16;
  int bcol = tn*16 + l16;
  const bf16*  Abf = (const bf16*)Aptr;
  const float* Af  = (const float*)Aptr;
  const bf16*  pbrow = Bt + (size_t)bcol*K;
  v8f acc = {};
  for (int k0 = 0; k0 < K; k0 += 32){
    v16bf a;
    if (AF32){
      const float* pa = Af + (size_t)arow*K + k0 + half*16;
      v8fv lo = *(const v8fv*)pa;
      v8fv hi = *(const v8fv*)(pa + 8);
#pragma unroll
      for (int i = 0; i < 8; ++i){ a[i] = (bf16)lo[i]; a[i+8] = (bf16)hi[i]; }
    } else {
      const bf16* pa = Abf + (size_t)arow*K + k0 + half*16;
      a = cat8(*(const v8bf*)pa, *(const v8bf*)(pa + 8));
    }
    v16bf bfr = *(const v16bf*)(pbrow + k0 + half*16);
    __builtin_prefetch((const void*)(pbrow + k0 + 64), 0, 1);
    acc = __builtin_amdgcn_wmma_f32_16x16x32_bf16(false, a, false, bfr, (short)0, acc, false, false);
  }
  float scal = scale_ptr ? scale_ptr[0] : 1.0f;
#pragma unroll
  for (int r = 0; r < 8; ++r){
    int row = tm*16 + r + half*8;
    size_t oi = (size_t)row*N + bcol;
    float v = acc[r] + (bias ? bias[bcol] : 0.0f);
    switch (epmode){
      case 0: out_f[oi] = v; break;
      case 1: out_bf[oi] = (bf16)(alpha*v); break;
      case 2: out_bf[oi] = (bf16)(addsrc[oi] + scal*v); break;
      case 3: out_bf[oi] = (bf16)gelu_f(v); break;
      case 4: out_f[oi] = addsrc[oi] + v; break;
    }
  }
}

// ---------------- 3x3 SAME conv as 9-tap WMMA GEMM (NHWC) ----------------
// mode 0: out_bf = bf16(gelu(acc)); mode 1: out_f = acc
__global__ void conv3x3_wmma_kernel(const bf16* __restrict__ in, const bf16* __restrict__ Wt,
                                    bf16* __restrict__ out_bf, float* __restrict__ out_f, int mode){
  int gw = (blockIdx.x*blockDim.x + threadIdx.x) >> 5;
  int lane = threadIdx.x & 31;
  const int tilesN = CB/16;                 // 12
  const int ntiles = (ROWS/16)*tilesN;      // 24576
  if (gw >= ntiles) return;
  int tm = gw / tilesN, tn = gw % tilesN;
  int half = lane >> 4, l16 = lane & 15;
  int p = tm*16 + l16;                      // spatial pos for this lane's A row
  int bimg = p >> 14;
  int rem = p & 16383;
  int yy = rem >> 7, xx = rem & 127;
  v8f acc = {};
  for (int tap = 0; tap < 9; ++tap){
    int dy = tap/3 - 1, dx = tap%3 - 1;
    int y2 = yy + dy, x2 = xx + dx;
    bool ok = ((unsigned)y2 < (unsigned)HH) && ((unsigned)x2 < (unsigned)WW);
    const bf16* rowp = in + (size_t)(bimg*NN + y2*WW + x2)*CB;
    const bf16* wt   = Wt + (size_t)tap*CB*CB + (size_t)(tn*16 + l16)*CB;
#pragma unroll
    for (int k0 = 0; k0 < CB; k0 += 32){
      v16bf a = {};
      if (ok){
        const bf16* pa = rowp + k0 + half*16;
        a = cat8(*(const v8bf*)pa, *(const v8bf*)(pa + 8));
      }
      v16bf bfr = *(const v16bf*)(wt + k0 + half*16);
      acc = __builtin_amdgcn_wmma_f32_16x16x32_bf16(false, a, false, bfr, (short)0, acc, false, false);
    }
  }
#pragma unroll
  for (int r = 0; r < 8; ++r){
    int row = tm*16 + r + half*8;
    int col = tn*16 + l16;
    size_t oi = (size_t)row*CB + col;
    if (mode == 0) out_bf[oi] = (bf16)gelu_f(acc[r]);
    else           out_f[oi] = acc[r];
  }
}

// ---------------- channel attention: pool + tiny MLP ----------------
__global__ void pool_kernel(const float* __restrict__ r, float* __restrict__ pooled){
  __shared__ float sm[256];
  int b = blockIdx.x / CB, c = blockIdx.x % CB;
  float s = 0.f;
  for (int n = threadIdx.x; n < NN; n += 256)
    s += r[((size_t)b*NN + n)*CB + c];
  sm[threadIdx.x] = s;
  __syncthreads();
  for (int off = 128; off > 0; off >>= 1){
    if (threadIdx.x < off) sm[threadIdx.x] += sm[threadIdx.x + off];
    __syncthreads();
  }
  if (threadIdx.x == 0) pooled[blockIdx.x] = sm[0] * (1.0f/NN);
}

__global__ void ca_mlp_kernel(const float* __restrict__ pooled, const float* __restrict__ w1,
                              const float* __restrict__ w2, float* __restrict__ y){
  __shared__ float t1[BB][12];
  int tid = threadIdx.x;
  if (tid < BB*12){
    int b = tid / 12, j = tid % 12;
    float s = 0.f;
    for (int c = 0; c < CB; ++c) s += w1[(size_t)j*CB + c]*pooled[b*CB + c];
    t1[b][j] = fmaxf(s, 0.f);
  }
  __syncthreads();
  if (tid < BB*CB){
    int b = tid / CB, c = tid % CB;
    float s = 0.f;
#pragma unroll
    for (int j = 0; j < 12; ++j) s += w2[(size_t)c*12 + j]*t1[b][j];
    y[tid] = 1.0f/(1.0f + expf(-s));
  }
}

// ---------------- windowed dual-KV differential attention core ----------------
__global__ void __launch_bounds__(128)
attn_kernel(const bf16* __restrict__ q_bf, const bf16* __restrict__ kvg,
            const bf16* __restrict__ kvs, const float* __restrict__ bias_tab,
            const float* __restrict__ lq1p, const float* __restrict__ lk1p,
            const float* __restrict__ lq2p, const float* __restrict__ lk2p,
            float* __restrict__ attn_raw){
  __shared__ __align__(32) bf16 vt[2][HD][NW2];       // [mat][d][key]
  __shared__ __align__(32) bf16 probs[4][16][NW2];    // per-wave softmax probs
  int blk = blockIdx.x;
  int w = blk / NHEAD, h = blk % NHEAD;
  int b  = w >> 8;
  int wy = (w >> 4) & 15;
  int wx = w & 15;
  int baserow = b*NN + wy*8*WW + wx*8;                 // row(t) = baserow + (t>>3)*128 + (t&7)
  int tid = threadIdx.x;

  for (int i = tid; i < NW2*HD; i += 128){
    int key = i >> 5, d = i & 31;
    int grow = baserow + (key >> 3)*WW + (key & 7);
    size_t off = (size_t)grow*(2*CB) + CB + h*HD + d;
    vt[0][d][key] = kvg[off];
    vt[1][d][key] = kvs[off];
  }
  __syncthreads();

  int wid = tid >> 5, lane = tid & 31, half = lane >> 4, l16 = lane & 15;
  int m0 = wid*16;

  int qm = m0 + l16;
  int qrow = baserow + (qm >> 3)*WW + (qm & 7);
  const bf16* qp = q_bf + (size_t)qrow*CB + h*HD + half*16;
  v16bf afq = cat8(*(const v8bf*)qp, *(const v8bf*)(qp + 8));

  float lam = expf(lq1p[0]*lk1p[0]) - expf(lq2p[0]*lk2p[0]) + LAMBDA_INIT;

  float outv[2][8];
#pragma unroll
  for (int nt = 0; nt < 2; ++nt)
#pragma unroll
    for (int r = 0; r < 8; ++r) outv[nt][r] = 0.f;

  for (int mat = 0; mat < 2; ++mat){
    const bf16* kv = (mat == 0) ? kvg : kvs;
    v8f sc[4];
#pragma unroll
    for (int tnn = 0; tnn < 4; ++tnn){
      int key = tnn*16 + l16;
      int grow = baserow + (key >> 3)*WW + (key & 7);
      const bf16* kp = kv + (size_t)grow*(2*CB) + h*HD + half*16;
      v16bf bfr = *(const v16bf*)kp;
      v8f z = {};
      sc[tnn] = __builtin_amdgcn_wmma_f32_16x16x32_bf16(false, afq, false, bfr, (short)0, z, false, false);
#pragma unroll
      for (int r = 0; r < 8; ++r){
        int srow = m0 + r + half*8;
        sc[tnn][r] += bias_tab[(size_t)h*NW2*NW2 + srow*NW2 + tnn*16 + l16];
      }
    }
    // softmax over keys per query row (rows live in 16-lane halves)
    float pv[4][8];
#pragma unroll
    for (int r = 0; r < 8; ++r){
      float mx = fmaxf(fmaxf(sc[0][r], sc[1][r]), fmaxf(sc[2][r], sc[3][r]));
#pragma unroll
      for (int m = 1; m < 16; m <<= 1) mx = fmaxf(mx, __shfl_xor(mx, m, 32));
      float sum = 0.f;
#pragma unroll
      for (int tnn = 0; tnn < 4; ++tnn){ float e = expf(sc[tnn][r] - mx); pv[tnn][r] = e; sum += e; }
#pragma unroll
      for (int m = 1; m < 16; m <<= 1) sum += __shfl_xor(sum, m, 32);
      float inv = 1.0f/sum;
#pragma unroll
      for (int tnn = 0; tnn < 4; ++tnn) pv[tnn][r] *= inv;
    }
    __syncthreads();   // previous out-GEMM reads of probs done
#pragma unroll
    for (int tnn = 0; tnn < 4; ++tnn)
#pragma unroll
      for (int r = 0; r < 8; ++r)
        probs[wid][r + half*8][tnn*16 + l16] = (bf16)pv[tnn][r];
    __syncthreads();
    // out = probs(16x64) @ v(64x32)
#pragma unroll
    for (int nt = 0; nt < 2; ++nt){
      v8f oacc = {};
#pragma unroll
      for (int ks = 0; ks < 2; ++ks){
        int k0 = ks*32;
        const bf16* pa = &probs[wid][l16][k0 + half*16];
        v16bf aa = cat8(*(const v8bf*)pa, *(const v8bf*)(pa + 8));
        const bf16* pvv = &vt[mat][nt*16 + l16][k0 + half*16];
        v16bf bb = *(const v16bf*)pvv;
        oacc = __builtin_amdgcn_wmma_f32_16x16x32_bf16(false, aa, false, bb, (short)0, oacc, false, false);
      }
#pragma unroll
      for (int r = 0; r < 8; ++r)
        outv[nt][r] += (mat == 0) ? oacc[r] : (-lam*oacc[r]);
    }
    __syncthreads();
  }
#pragma unroll
  for (int nt = 0; nt < 2; ++nt)
#pragma unroll
    for (int r = 0; r < 8; ++r){
      int qm2 = m0 + r + half*8;
      int grow = baserow + (qm2 >> 3)*WW + (qm2 & 7);
      attn_raw[(size_t)grow*CB + h*HD + nt*16 + l16] = outv[nt][r];
    }
}

// ---------------- residual combine: h = x + attn_out + 0.01*(r*y + xn) ----------------
__global__ void combine_kernel(const float* __restrict__ x, const float* __restrict__ attn_out,
                               const float* __restrict__ r, const float* __restrict__ y,
                               const float* __restrict__ xn, float* __restrict__ h){
  size_t idx = (size_t)blockIdx.x*blockDim.x + threadIdx.x;
  if (idx >= (size_t)ROWS*CB) return;
  int c = (int)(idx % CB);
  int b = (int)((idx / CB) >> 14);
  h[idx] = x[idx] + attn_out[idx] + 0.01f*(r[idx]*y[b*CB + c] + xn[idx]);
}

// ---------------- launch ----------------
static inline unsigned cdivu(long a, long b){ return (unsigned)((a + b - 1)/b); }

extern "C" void kernel_launch(void* const* d_in, const int* in_sizes, int n_in,
                              void* d_out, int out_size, void* d_ws, size_t ws_size,
                              hipStream_t stream){
  const float* x      = (const float*)d_in[0];
  const float* dino   = (const float*)d_in[1];
  const float* pf     = (const float*)d_in[2];
  const float* ln1_g  = (const float*)d_in[3];
  const float* ln1_b  = (const float*)d_in[4];
  const float* Wq     = (const float*)d_in[5];
  const float* bq     = (const float*)d_in[6];
  const float* Wkvg   = (const float*)d_in[7];
  const float* bkvg   = (const float*)d_in[8];
  const float* Wkvs   = (const float*)d_in[9];
  const float* bkvs   = (const float*)d_in[10];
  const float* Wgeo   = (const float*)d_in[11];
  const float* bgeo   = (const float*)d_in[12];
  const float* Wdino  = (const float*)d_in[13];
  const float* bdino  = (const float*)d_in[14];
  const float* geo_w  = (const float*)d_in[15];
  const float* sem_w  = (const float*)d_in[16];
  const float* lq1    = (const float*)d_in[17];
  const float* lk1    = (const float*)d_in[18];
  const float* lq2    = (const float*)d_in[19];
  const float* lk2    = (const float*)d_in[20];
  const float* rpb    = (const float*)d_in[21];
  const float* subln_g= (const float*)d_in[22];
  const float* subln_b= (const float*)d_in[23];
  const float* Wproj  = (const float*)d_in[24];
  const float* bproj  = (const float*)d_in[25];
  const float* cab_w1 = (const float*)d_in[26];
  const float* cab_w2 = (const float*)d_in[27];
  const float* ca_w1  = (const float*)d_in[28];
  const float* ca_w2  = (const float*)d_in[29];
  const float* ln2_g  = (const float*)d_in[30];
  const float* ln2_b  = (const float*)d_in[31];
  const float* Wm1    = (const float*)d_in[32];
  const float* bm1    = (const float*)d_in[33];
  const float* Wm2    = (const float*)d_in[34];
  const float* bm2    = (const float*)d_in[35];

  char* ws = (char*)d_ws;
  size_t off = 0;
  auto alloc = [&](size_t bytes)->char*{
    char* p = ws + off;
    off = (off + bytes + 255) & ~(size_t)255;
    return p;
  };
  const size_t NC_F = (size_t)ROWS*CB*sizeof(float);   // 25.2 MB
  const size_t NC_B = (size_t)ROWS*CB*sizeof(bf16);    // 12.6 MB

  float* xn        = (float*)alloc(NC_F);
  bf16*  xn_bf     = (bf16*) alloc(NC_B);
  bf16*  q_bf      = (bf16*) alloc(NC_B);
  bf16*  geoe_bf   = (bf16*) alloc(NC_B);
  bf16*  seme_bf   = (bf16*) alloc(NC_B);
  bf16*  kvg_bf    = (bf16*) alloc(2*NC_B);
  bf16*  kvs_bf    = (bf16*) alloc(2*NC_B);
  float* attn_raw  = (float*)alloc(NC_F);
  bf16*  attnln_bf = (bf16*) alloc(NC_B);
  float* attn_out  = (float*)alloc(NC_F);
  bf16*  conv1_bf  = (bf16*) alloc(NC_B);
  float* r_f32     = (float*)alloc(NC_F);
  float* h_f32     = (float*)alloc(NC_F);
  bf16*  hln_bf    = (bf16*) alloc(NC_B);
  bf16*  mlp1_bf   = (bf16*) alloc((size_t)ROWS*MLPH*sizeof(bf16));
  float* pooled    = (float*)alloc(BB*CB*sizeof(float));
  float* y_ca      = (float*)alloc(BB*CB*sizeof(float));
  float* bias_tab  = (float*)alloc((size_t)NHEAD*NW2*NW2*sizeof(float));
  bf16*  Wq_t      = (bf16*) alloc((size_t)CB*CB*sizeof(bf16));
  bf16*  Wkvg_t    = (bf16*) alloc((size_t)CB*2*CB*sizeof(bf16));
  bf16*  Wkvs_t    = (bf16*) alloc((size_t)CB*2*CB*sizeof(bf16));
  bf16*  Wdino_t   = (bf16*) alloc((size_t)1024*CB*sizeof(bf16));
  bf16*  Wproj_t   = (bf16*) alloc((size_t)CB*CB*sizeof(bf16));
  bf16*  Wm1_t     = (bf16*) alloc((size_t)CB*MLPH*sizeof(bf16));
  bf16*  Wm2_t     = (bf16*) alloc((size_t)MLPH*CB*sizeof(bf16));
  bf16*  cw1_t     = (bf16*) alloc((size_t)9*CB*CB*sizeof(bf16));
  bf16*  cw2_t     = (bf16*) alloc((size_t)9*CB*CB*sizeof(bf16));
  (void)ws_size; (void)in_sizes; (void)n_in; (void)out_size;

  // ---- weight prep ----
  transpose_bf_kernel<<<cdivu(CB*CB,256),256,0,stream>>>(Wq, Wq_t, CB, CB);
  transpose_bf_kernel<<<cdivu(CB*2*CB,256),256,0,stream>>>(Wkvg, Wkvg_t, CB, 2*CB);
  transpose_bf_kernel<<<cdivu(CB*2*CB,256),256,0,stream>>>(Wkvs, Wkvs_t, CB, 2*CB);
  transpose_bf_kernel<<<cdivu(1024*CB,256),256,0,stream>>>(Wdino, Wdino_t, 1024, CB);
  transpose_bf_kernel<<<cdivu(CB*CB,256),256,0,stream>>>(Wproj, Wproj_t, CB, CB);
  transpose_bf_kernel<<<cdivu(CB*MLPH,256),256,0,stream>>>(Wm1, Wm1_t, CB, MLPH);
  transpose_bf_kernel<<<cdivu(MLPH*CB,256),256,0,stream>>>(Wm2, Wm2_t, MLPH, CB);
  convw_prep_kernel<<<cdivu(9*CB*CB,256),256,0,stream>>>(cab_w1, cw1_t);
  convw_prep_kernel<<<cdivu(9*CB*CB,256),256,0,stream>>>(cab_w2, cw2_t);
  biastab_kernel<<<cdivu(NHEAD*NW2*NW2,256),256,0,stream>>>(rpb, bias_tab);

  // ---- LN1 ----
  layernorm_kernel<<<cdivu((long)ROWS*32,256),256,0,stream>>>(x, ln1_g, ln1_b, xn, xn_bf, 1.0f, ROWS);

  // ---- CAB conv branch ----
  {
    unsigned blocks = cdivu((long)(ROWS/16)*(CB/16), 8);
    conv3x3_wmma_kernel<<<blocks,256,0,stream>>>(xn_bf, cw1_t, conv1_bf, nullptr, 0);
    conv3x3_wmma_kernel<<<blocks,256,0,stream>>>(conv1_bf, cw2_t, nullptr, r_f32, 1);
  }
  pool_kernel<<<BB*CB,256,0,stream>>>(r_f32, pooled);
  ca_mlp_kernel<<<1,BB*CB,0,stream>>>(pooled, ca_w1, ca_w2, y_ca);

  // ---- attention projections (WMMA GEMMs) ----
  auto gemm_tiles = [](int M, int N){ return (unsigned)(((long)(M/16)*(N/16) + 7)/8); };
  // q = (xn @ Wq + bq) * HD^-0.5 -> bf16
  gemm_wmma_kernel<0><<<gemm_tiles(ROWS,CB),256,0,stream>>>(
      xn_bf, Wq_t, bq, nullptr, nullptr, q_bf, nullptr, ROWS, CB, CB, 1, QSCALE);
  // geo_e (K=3 elementwise)
  geo_kernel<<<cdivu((long)ROWS*CB,256),256,0,stream>>>(xn, pf, Wgeo, bgeo, geo_w, geoe_bf);
  // sem_e = xn + sem_w*(dino @ Wdino + bdino) -> bf16 (A is f32, converted on the fly)
  gemm_wmma_kernel<1><<<gemm_tiles(ROWS,CB),256,0,stream>>>(
      dino, Wdino_t, bdino, xn, sem_w, seme_bf, nullptr, ROWS, 1024, CB, 2, 1.0f);
  // kv projections
  gemm_wmma_kernel<0><<<gemm_tiles(ROWS,2*CB),256,0,stream>>>(
      geoe_bf, Wkvg_t, bkvg, nullptr, nullptr, kvg_bf, nullptr, ROWS, CB, 2*CB, 1, 1.0f);
  gemm_wmma_kernel<0><<<gemm_tiles(ROWS,2*CB),256,0,stream>>>(
      seme_bf, Wkvs_t, bkvs, nullptr, nullptr, kvs_bf, nullptr, ROWS, CB, 2*CB, 1, 1.0f);

  // ---- attention core ----
  attn_kernel<<<(BB*256)*NHEAD,128,0,stream>>>(q_bf, kvg_bf, kvs_bf, bias_tab,
                                               lq1, lk1, lq2, lk2, attn_raw);

  // ---- subLN * (1 - lambda_init), then proj ----
  layernorm_kernel<<<cdivu((long)ROWS*32,256),256,0,stream>>>(
      attn_raw, subln_g, subln_b, nullptr, attnln_bf, 1.0f - LAMBDA_INIT, ROWS);
  gemm_wmma_kernel<0><<<gemm_tiles(ROWS,CB),256,0,stream>>>(
      attnln_bf, Wproj_t, bproj, nullptr, nullptr, nullptr, attn_out, ROWS, CB, CB, 0, 1.0f);

  // ---- residual combine + MLP ----
  combine_kernel<<<cdivu((long)ROWS*CB,256),256,0,stream>>>(x, attn_out, r_f32, y_ca, xn, h_f32);
  layernorm_kernel<<<cdivu((long)ROWS*32,256),256,0,stream>>>(h_f32, ln2_g, ln2_b, nullptr, hln_bf, 1.0f, ROWS);
  gemm_wmma_kernel<0><<<gemm_tiles(ROWS,MLPH),256,0,stream>>>(
      hln_bf, Wm1_t, bm1, nullptr, nullptr, mlp1_bf, nullptr, ROWS, CB, MLPH, 3, 1.0f);
  gemm_wmma_kernel<0><<<gemm_tiles(ROWS,CB),256,0,stream>>>(
      mlp1_bf, Wm2_t, bm2, h_f32, nullptr, nullptr, (float*)d_out, ROWS, MLPH, CB, 4, 1.0f);
}
